// Transducer_42949672960725
// MI455X (gfx1250) — compile-verified, matrix operand order checked by reading, not run on hardware
//
#include <hip/hip_runtime.h>
#include <hip/hip_bf16.h>

typedef __attribute__((ext_vector_type(16))) __bf16 v16bf;
typedef __attribute__((ext_vector_type(8)))  float  v8f;

#define B_   8
#define T_   200
#define U1_  81     // U+1
#define H_   512
#define E_   256
#define NV_  513    // V+1
#define NPAD_ 544   // 34 WMMA n-tiles of 16
#define KSTEPS_ 16  // 512 / 32

// ---------------------------------------------------------------------------
// Pack fc2_w (513x512 fp32) -> bf16, K-blocked [ks][v_pad][kk], v padded to 544
// ---------------------------------------------------------------------------
__global__ __launch_bounds__(256) void pack_fc2(const float* __restrict__ w,
                                                __bf16* __restrict__ dst) {
    int idx = blockIdx.x * 256 + threadIdx.x;
    const int total = KSTEPS_ * NPAD_ * 32;
    if (idx >= total) return;
    int ks  = idx / (NPAD_ * 32);
    int rem = idx % (NPAD_ * 32);
    int v   = rem >> 5;
    int kk  = rem & 31;
    float val = (v < NV_) ? w[(size_t)v * H_ + ks * 32 + kk] : 0.f;
    dst[idx] = (__bf16)val;
}

__global__ __launch_bounds__(256) void zero_f(float* __restrict__ p, int n) {
    int i = blockIdx.x * 256 + threadIdx.x;
    if (i < n) p[i] = 0.f;
}

// ---------------------------------------------------------------------------
// Embedding gather + blank start token: ycat (B, U1, E)
// ---------------------------------------------------------------------------
__global__ __launch_bounds__(E_) void gather_embed(const int* __restrict__ y,
                                                   const float* __restrict__ emb,
                                                   float* __restrict__ ycat) {
    int b = blockIdx.x, u = blockIdx.y, tid = threadIdx.x;
    float v = 0.f;
    if (u > 0) {
        int id = y[b * (U1_ - 1) + (u - 1)];
        v = emb[(size_t)id * E_ + tid];
    }
    ycat[((size_t)b * U1_ + u) * E_ + tid] = v;
}

// ---------------------------------------------------------------------------
// Generic small GEMM: C(M,N) = A(M,K) * W(N,K)^T + bias(N)   (fp32, L2-resident)
// ---------------------------------------------------------------------------
__global__ __launch_bounds__(256) void gemm_nt_bias(const float* __restrict__ A,
                                                    const float* __restrict__ W,
                                                    const float* __restrict__ bias,
                                                    float* __restrict__ C,
                                                    int M, int N, int K) {
    int idx = blockIdx.x * 256 + threadIdx.x;
    if (idx >= M * N) return;
    int m = idx / N, n = idx % N;
    const float4* a = (const float4*)(A + (size_t)m * K);
    const float4* w = (const float4*)(W + (size_t)n * K);
    float s = 0.f;
    int k4 = K >> 2;
    for (int k = 0; k < k4; ++k) {
        float4 av = a[k], wv = w[k];
        s += av.x * wv.x + av.y * wv.y + av.z * wv.z + av.w * wv.w;
    }
    C[idx] = s + bias[n];
}

// ---------------------------------------------------------------------------
// One GRU time step: h_new = gate(x_proj[:,u,:], h_prev @ w_hh^T + b_hh)
// grid (B, 4) x 128 threads; each thread computes one hidden column.
// ---------------------------------------------------------------------------
__global__ __launch_bounds__(128) void gru_step(const float* __restrict__ xproj,
                                                const float* __restrict__ whh,
                                                const float* __restrict__ bhh,
                                                const float* __restrict__ hprev,
                                                float* __restrict__ hnext,
                                                float* __restrict__ dec, int u) {
    int b = blockIdx.x;
    int i = blockIdx.y * 128 + threadIdx.x;      // 0..511
    const float4* hv = (const float4*)(hprev + (size_t)b * H_);
    const float4* wr = (const float4*)(whh + (size_t)i * H_);
    const float4* wz = (const float4*)(whh + (size_t)(H_ + i) * H_);
    const float4* wn = (const float4*)(whh + (size_t)(2 * H_ + i) * H_);
    float sr = 0.f, sz = 0.f, sn = 0.f;
    for (int k = 0; k < H_ / 4; ++k) {
        float4 h4 = hv[k];
        float4 r4 = wr[k], z4 = wz[k], n4 = wn[k];
        sr += h4.x * r4.x + h4.y * r4.y + h4.z * r4.z + h4.w * r4.w;
        sz += h4.x * z4.x + h4.y * z4.y + h4.z * z4.z + h4.w * z4.w;
        sn += h4.x * n4.x + h4.y * n4.y + h4.z * n4.z + h4.w * n4.w;
    }
    const float* xp = xproj + ((size_t)b * U1_ + u) * (3 * H_);
    float xr = xp[i], xz = xp[H_ + i], xn = xp[2 * H_ + i];
    float hr = sr + bhh[i];
    float hz = sz + bhh[H_ + i];
    float hn = sn + bhh[2 * H_ + i];
    float r = 1.f / (1.f + __expf(-(xr + hr)));
    float z = 1.f / (1.f + __expf(-(xz + hz)));
    float n = tanhf(xn + r * hn);
    float hp = hprev[(size_t)b * H_ + i];
    float h = (1.f - z) * n + z * hp;
    hnext[(size_t)b * H_ + i] = h;
    dec[((size_t)b * U1_ + u) * H_ + i] = h;
}

// ---------------------------------------------------------------------------
// Joint + fc2 GEMM via v_wmma_f32_16x16x32_bf16, fused with log_softmax.
//  Each block: M=64 rows of joint (rows = (b,t,u)), N = 544 (513 padded).
//  8 waves: wave = (m_sub 0..3, n_half 0..1); each wave owns 17 n-tiles.
//  A fragment built per-lane in registers: bf16(relu(fx[b,t,:] + fy[b,u,:])).
//  B slab double-buffered in LDS via global_load_async_to_lds_b128;
//  copy of step ks+1 overlaps the 17 WMMAs of step ks (one barrier/step).
//  Epilogue: bias fold + row-wise log_softmax, single write of the output.
// ---------------------------------------------------------------------------
#define ROWB 48  // LDS row stride in __bf16 elems (96 bytes)

__global__ __launch_bounds__(256) void joint_fc2_wmma(const float* __restrict__ fx,
                                                      const float* __restrict__ fy,
                                                      const __bf16* __restrict__ wsBt,
                                                      const float* __restrict__ fc2b,
                                                      float* __restrict__ out) {
    __shared__ __align__(64) __bf16 ldsB[2][NPAD_ * ROWB];  // 2 x 52224 B
    __shared__ float sredM[2][64];
    __shared__ float sredS[2][64];

    const int tid    = threadIdx.x;
    const int wave   = tid >> 5;
    const int lane   = tid & 31;
    const int laneLo = lane & 15;
    const int laneHi = lane >> 4;
    const int m_sub  = wave >> 1;   // 0..3
    const int n_half = wave & 1;    // 0..1
    const int rbase  = blockIdx.x * 64;

    // per-lane A row -> (b,t,u) -> fx/fy base offsets (computed once)
    int myrow = rbase + m_sub * 16 + laneLo;
    int bb  = myrow / (T_ * U1_);
    int rem = myrow % (T_ * U1_);
    int tt  = rem / U1_;
    int uu  = rem % U1_;
    const float* pfx = fx + (size_t)(bb * T_ + tt) * H_ + laneHi * 16;
    const float* pfy = fy + (size_t)(bb * U1_ + uu) * H_ + laneHi * 16;

    v8f acc[17];
    const v8f vzero = {0.f, 0.f, 0.f, 0.f, 0.f, 0.f, 0.f, 0.f};
#pragma unroll
    for (int j = 0; j < 17; ++j) acc[j] = vzero;

    const unsigned ldsB0 = (unsigned)(size_t)(&ldsB[0][0]);
    const unsigned bufStride = (unsigned)(NPAD_ * ROWB * 2);  // bytes per buffer

    // ---- async B-slab copy: 2176 x 16B chunks, 96B LDS row stride -------
    auto issueB = [&](int ks, int buf) {
        unsigned long long slab =
            (unsigned long long)(size_t)(wsBt + (size_t)ks * (NPAD_ * 32));
        unsigned base = ldsB0 + (unsigned)buf * bufStride;
#pragma unroll
        for (int c = 0; c < 9; ++c) {
            int idx = tid + c * 256;
            if (idx < NPAD_ * 32 / 8) {
                int row = idx >> 2, ch = idx & 3;
                unsigned ldsOff = base + (unsigned)(row * 96 + ch * 16);
                unsigned gOff = (unsigned)idx * 16u;
                asm volatile("global_load_async_to_lds_b128 %0, %1, %2 offset:0"
                             :: "v"(ldsOff), "v"(gOff), "s"(slab)
                             : "memory");
            }
        }
    };

    // prologue: start copy of slab 0
    issueB(0, 0);

    for (int ks = 0; ks < KSTEPS_; ++ks) {
        const int cur = ks & 1;
        // drain this wave's outstanding asyncs (slab ks), then block barrier
        asm volatile("s_wait_asynccnt 0" ::: "memory");
        __syncthreads();
        // kick off next slab copy; overlaps with the WMMAs below
        if (ks < KSTEPS_ - 1) issueB(ks + 1, cur ^ 1);

        // ---- build A fragment in registers: relu(fx+fy) -> bf16 --------
        union { __bf16 h[16]; v16bf v; } af;
#pragma unroll
        for (int q = 0; q < 4; ++q) {
            float4 xa = ((const float4*)(pfx + ks * 32))[q];
            float4 ya = ((const float4*)(pfy + ks * 32))[q];
            float s0 = xa.x + ya.x, s1 = xa.y + ya.y;
            float s2 = xa.z + ya.z, s3 = xa.w + ya.w;
            af.h[q * 4 + 0] = (__bf16)(s0 > 0.f ? s0 : 0.f);
            af.h[q * 4 + 1] = (__bf16)(s1 > 0.f ? s1 : 0.f);
            af.h[q * 4 + 2] = (__bf16)(s2 > 0.f ? s2 : 0.f);
            af.h[q * 4 + 3] = (__bf16)(s3 > 0.f ? s3 : 0.f);
        }
        v16bf afrag = af.v;

        // ---- 17 WMMAs on current B buffer, 1-deep bfrag pipeline --------
        const __bf16* bbuf = &ldsB[cur][0];
        v16bf bfrag = *(const v16bf*)(bbuf + ((n_half * 17) * 16 + laneLo) * ROWB + laneHi * 16);
#pragma unroll
        for (int j = 0; j < 17; ++j) {
            v16bf bnext;
            if (j < 16) {
                int vrow = (n_half * 17 + j + 1) * 16 + laneLo;
                bnext = *(const v16bf*)(bbuf + vrow * ROWB + laneHi * 16);
            }
            acc[j] = __builtin_amdgcn_wmma_f32_16x16x32_bf16(
                false, afrag, false, bfrag, (short)0, acc[j], false, false);
            bfrag = bnext;
        }
    }
    __syncthreads();

    // ---- fold bias into accumulators (padded cols get 0, masked later) --
#pragma unroll
    for (int j = 0; j < 17; ++j) {
        int col = (n_half * 17 + j) * 16 + laneLo;
        float bias = (col < NV_) ? fc2b[col] : 0.f;
#pragma unroll
        for (int i = 0; i < 8; ++i) acc[j][i] += bias;
    }

    // ---- row max: per-lane over valid tiles, then shfl over 16 lanes ----
    float lmax[8];
#pragma unroll
    for (int i = 0; i < 8; ++i) lmax[i] = -3.0e38f;
#pragma unroll
    for (int j = 0; j < 17; ++j) {
        int col = (n_half * 17 + j) * 16 + laneLo;
        if (col < NV_) {
#pragma unroll
            for (int i = 0; i < 8; ++i) lmax[i] = fmaxf(lmax[i], acc[j][i]);
        }
    }
#pragma unroll
    for (int i = 0; i < 8; ++i) {
        for (int d = 1; d < 16; d <<= 1)
            lmax[i] = fmaxf(lmax[i], __shfl_xor(lmax[i], d, 32));
    }
    const int rowloc = m_sub * 16 + laneHi * 8;
    if (laneLo == 0) {
#pragma unroll
        for (int i = 0; i < 8; ++i) sredM[n_half][rowloc + i] = lmax[i];
    }
    __syncthreads();
#pragma unroll
    for (int i = 0; i < 8; ++i)
        lmax[i] = fmaxf(sredM[0][rowloc + i], sredM[1][rowloc + i]);

    // ---- sum of exp -----------------------------------------------------
    float lsum[8];
#pragma unroll
    for (int i = 0; i < 8; ++i) lsum[i] = 0.f;
#pragma unroll
    for (int j = 0; j < 17; ++j) {
        int col = (n_half * 17 + j) * 16 + laneLo;
        if (col < NV_) {
#pragma unroll
            for (int i = 0; i < 8; ++i) lsum[i] += __expf(acc[j][i] - lmax[i]);
        }
    }
#pragma unroll
    for (int i = 0; i < 8; ++i) {
        for (int d = 1; d < 16; d <<= 1)
            lsum[i] += __shfl_xor(lsum[i], d, 32);
    }
    if (laneLo == 0) {
#pragma unroll
        for (int i = 0; i < 8; ++i) sredS[n_half][rowloc + i] = lsum[i];
    }
    __syncthreads();
    float lse[8];
#pragma unroll
    for (int i = 0; i < 8; ++i)
        lse[i] = lmax[i] + __logf(sredS[0][rowloc + i] + sredS[1][rowloc + i]);

    // ---- single store of normalized log-probs ---------------------------
#pragma unroll
    for (int j = 0; j < 17; ++j) {
        int col = (n_half * 17 + j) * 16 + laneLo;
        if (col < NV_) {
#pragma unroll
            for (int i = 0; i < 8; ++i) {
                int row = rbase + m_sub * 16 + laneHi * 8 + i;
                out[(size_t)row * NV_ + col] = acc[j][i] - lse[i];
            }
        }
    }
}

// ---------------------------------------------------------------------------
extern "C" void kernel_launch(void* const* d_in, const int* in_sizes, int n_in,
                              void* d_out, int out_size, void* d_ws, size_t ws_size,
                              hipStream_t stream) {
    const float* x    = (const float*)d_in[0];
    const int*   y    = (const int*)  d_in[1];
    const float* emb  = (const float*)d_in[2];
    const float* wih  = (const float*)d_in[3];
    const float* whh  = (const float*)d_in[4];
    const float* bih  = (const float*)d_in[5];
    const float* bhh  = (const float*)d_in[6];
    const float* fc1w = (const float*)d_in[7];
    const float* fc1b = (const float*)d_in[8];
    const float* fc2w = (const float*)d_in[9];
    const float* fc2b = (const float*)d_in[10];
    float* out = (float*)d_out;

    char* wsp = (char*)d_ws;
    auto alloc = [&](size_t bytes) {
        char* p = wsp;
        wsp += (bytes + 255) & ~(size_t)255;
        return p;
    };
    float*  ycat  = (float*)alloc((size_t)B_ * U1_ * E_ * 4);
    float*  xproj = (float*)alloc((size_t)B_ * U1_ * 3 * H_ * 4);
    float*  hbuf  = (float*)alloc((size_t)2 * B_ * H_ * 4);
    float*  dec   = (float*)alloc((size_t)B_ * U1_ * H_ * 4);
    float*  fxbuf = (float*)alloc((size_t)B_ * T_ * H_ * 4);
    float*  fybuf = (float*)alloc((size_t)B_ * U1_ * H_ * 4);
    __bf16* wsBt  = (__bf16*)alloc((size_t)KSTEPS_ * NPAD_ * 32 * 2);

    // Pack fc2 weights -> bf16 K-blocked
    pack_fc2<<<(KSTEPS_ * NPAD_ * 32 + 255) / 256, 256, 0, stream>>>(fc2w, wsBt);
    // Zero GRU state ping-pong
    zero_f<<<(2 * B_ * H_ + 255) / 256, 256, 0, stream>>>(hbuf, 2 * B_ * H_);
    // Embedding gather + blank start
    gather_embed<<<dim3(B_, U1_), E_, 0, stream>>>(y, emb, ycat);
    // Precompute input gate projections: x_proj = ycat @ w_ih^T + b_ih
    gemm_nt_bias<<<(B_ * U1_ * 3 * H_ + 255) / 256, 256, 0, stream>>>(
        ycat, wih, bih, xproj, B_ * U1_, 3 * H_, E_);
    // Sequential GRU scan (81 graph-captured step kernels)
    for (int u = 0; u < U1_; ++u) {
        const float* hp = hbuf + (size_t)(u & 1) * (B_ * H_);
        float*       hn = hbuf + (size_t)((u + 1) & 1) * (B_ * H_);
        gru_step<<<dim3(B_, H_ / 128), 128, 0, stream>>>(xproj, whh, bhh, hp, hn, dec, u);
    }
    // fx = x @ fc1^T + b ; fy = dec @ fc1^T + b
    gemm_nt_bias<<<(B_ * T_ * H_ + 255) / 256, 256, 0, stream>>>(
        x, fc1w, fc1b, fxbuf, B_ * T_, H_, H_);
    gemm_nt_bias<<<(B_ * U1_ * H_ + 255) / 256, 256, 0, stream>>>(
        dec, fc1w, fc1b, fybuf, B_ * U1_, H_, H_);
    // Fused joint construction + fc2 GEMM (bf16 WMMA) + bias + log_softmax
    joint_fc2_wmma<<<(B_ * T_ * U1_) / 64, 256, 0, stream>>>(fxbuf, fybuf, wsBt, fc2b, out);
}
